// GNNEncoder_88510686035980
// MI455X (gfx1250) — matrix operand order repaired and checked by reading, not compiled, not run on hardware
//
#include <hip/hip_runtime.h>
#include <hip/hip_bf16.h>

typedef __attribute__((ext_vector_type(16))) _Float16 v16h;
typedef __attribute__((ext_vector_type(8)))  float    v8f;
typedef __attribute__((ext_vector_type(4)))  float    v4f;

#define H_DIM 128
#define BN_EPS 1e-5f
#define LK_SLOPE 0.01f

// ---------------------------------------------------------------------------
// Degree / normalization
// ---------------------------------------------------------------------------
__global__ void gcn_deg_init(float* __restrict__ deg, int n) {
    int i = blockIdx.x * blockDim.x + threadIdx.x;
    if (i < n) deg[i] = 1.0f;   // self-loop weight
}

__global__ void gcn_deg_accum(const int* __restrict__ ei, const float* __restrict__ w,
                              float* __restrict__ deg, int E) {
    int e = blockIdx.x * blockDim.x + threadIdx.x;
    if (e < E) atomicAdd(&deg[ei[E + e]], w[e]);   // dst = ei[1][e]
}

__global__ void gcn_deg_rsqrt(float* __restrict__ deg, int n) {
    int i = blockIdx.x * blockDim.x + threadIdx.x;
    if (i < n) deg[i] = rsqrtf(deg[i]);            // deg >= 1 always (self loop)
}

// ---------------------------------------------------------------------------
// Weight convert: f32 [in][out] row-major -> f16 [out][in]  (B-fragment friendly)
// ---------------------------------------------------------------------------
__global__ void gcn_wcvt(const float* __restrict__ W, _Float16* __restrict__ Wt) {
    int i = blockIdx.x * blockDim.x + threadIdx.x;   // 16384
    if (i < H_DIM * H_DIM) {
        int in = i >> 7, out = i & 127;
        Wt[out * H_DIM + in] = (_Float16)W[in * H_DIM + out];
    }
}

// ---------------------------------------------------------------------------
// WMMA GEMM:  t[N,128] = act[N,128] @ W  (W async-DMA'd into LDS as f16 [out][in])
// 128 threads = 4 waves; each wave -> 16 rows x 128 cols; block -> 64 rows.
// ---------------------------------------------------------------------------
__global__ void __launch_bounds__(128)
gcn_gemm_wmma(const float* __restrict__ act, const _Float16* __restrict__ Wt,
              float* __restrict__ t, int nrows) {
    __shared__ _Float16 lds_w[H_DIM * H_DIM];   // 32 KB of the 320 KB/WGP

    const int tid = threadIdx.x;

    // Async global->LDS staging of the 32KB weight tile (no VGPR round trip).
    // GVS mode: SADDR = 64-bit base, VADDR = per-lane 32-bit byte offset,
    // VDST VGPR = per-lane LDS byte offset. 2048 x 16B, 128 lanes, 16 iters.
    {
        const unsigned long long wbase = (unsigned long long)Wt;
        const unsigned ldsBase = (unsigned)(size_t)(&lds_w[0]);
#pragma unroll
        for (int i = 0; i < 16; ++i) {
            unsigned goff = (unsigned)(tid * 16 + i * 128 * 16);
            unsigned loff = ldsBase + goff;
            asm volatile("global_load_async_to_lds_b128 %0, %1, %2"
                         :
                         : "v"(loff), "v"(goff), "s"(wbase)
                         : "memory");
        }
        asm volatile("s_wait_asynccnt 0" ::: "memory");
    }
    __syncthreads();

    const int wave  = tid >> 5;
    const int lane  = tid & 31;
    const int hlf   = lane >> 4;
    const int mlan  = lane & 15;
    const int rowBase = blockIdx.x * 64 + wave * 16;

    v8f c[8] = {};   // 8 N-tiles of 16x16 f32 accumulators

#pragma unroll
    for (int ks = 0; ks < 4; ++ks) {             // K = 128 in steps of 32
        int row = rowBase + mlan;
        int r   = row < nrows ? row : nrows - 1; // clamp (stores are guarded)
        const float* ap = act + (size_t)r * H_DIM + ks * 32;
        // A fragment: elems 0..15 <- K=hlf*16+0..15 (contiguous, matches B)
        v4f a0 = *(const v4f*)(ap + hlf * 16);
        v4f a1 = *(const v4f*)(ap + hlf * 16 + 4);
        v4f a2 = *(const v4f*)(ap + hlf * 16 + 8);
        v4f a3 = *(const v4f*)(ap + hlf * 16 + 12);
        v16h a;
#pragma unroll
        for (int j = 0; j < 4; ++j) {
            a[j]      = (_Float16)a0[j];
            a[4 + j]  = (_Float16)a1[j];
            a[8 + j]  = (_Float16)a2[j];
            a[12 + j] = (_Float16)a3[j];
        }
#pragma unroll
        for (int nt = 0; nt < 8; ++nt) {
            int n = nt * 16 + mlan;
            // B fragment: lanes 0-15 hold K=ks*32+0..15, lanes 16-31 K=ks*32+16..31
            v16h b = *(const v16h*)(lds_w + n * H_DIM + ks * 32 + hlf * 16);
            c[nt] = __builtin_amdgcn_wmma_f32_16x16x32_f16(
                        false, a, false, b, (short)0, c[nt], false, false);
        }
    }

    // C/D layout: VGPR r -> row (r + 8*hlf), col = mlan within tile
#pragma unroll
    for (int nt = 0; nt < 8; ++nt) {
#pragma unroll
        for (int r = 0; r < 8; ++r) {
            int row = rowBase + r + hlf * 8;
            if (row < nrows)
                t[(size_t)row * H_DIM + nt * 16 + mlan] = c[nt][r];
        }
    }
}

// ---------------------------------------------------------------------------
// out[n,:] = t[n,:] * dinv[n]^2 + bias        (self-loop term + bias)
// 32 threads per node, 4 floats per thread
// ---------------------------------------------------------------------------
__global__ void gcn_selfloop_bias(const float* __restrict__ t, const float* __restrict__ dinv,
                                  const float* __restrict__ bias, float* __restrict__ out, int n) {
    int gid  = blockIdx.x * blockDim.x + threadIdx.x;
    int node = gid >> 5;
    int c    = (gid & 31) << 2;
    if (node < n) {
        float s  = dinv[node]; s = s * s;
        v4f tv = *(const v4f*)(t + (size_t)node * H_DIM + c);
        v4f bv = *(const v4f*)(bias + c);
        v4f o;
#pragma unroll
        for (int k = 0; k < 4; ++k) o[k] = tv[k] * s + bv[k];
        *(v4f*)(out + (size_t)node * H_DIM + c) = o;
    }
}

// ---------------------------------------------------------------------------
// Edge scatter: out[dst] += t[src] * (dinv[src]*w*dinv[dst])
// One wave per edge (all 32 lanes share e), float4 per lane, f32 L2 atomics
// ---------------------------------------------------------------------------
__global__ void gcn_scatter(const int* __restrict__ ei, const float* __restrict__ w,
                            const float* __restrict__ dinv, const float* __restrict__ t,
                            float* __restrict__ out, int E) {
    int gid  = blockIdx.x * blockDim.x + threadIdx.x;
    int e    = gid >> 5;
    int c    = (gid & 31) << 2;
    if (e < E) {
        int s = ei[e];
        int d = ei[E + e];
        float coef = dinv[s] * w[e] * dinv[d];
        v4f tv = *(const v4f*)(t + (size_t)s * H_DIM + c);
        float* od = out + (size_t)d * H_DIM + c;
        atomicAdd(od + 0, tv[0] * coef);
        atomicAdd(od + 1, tv[1] * coef);
        atomicAdd(od + 2, tv[2] * coef);
        atomicAdd(od + 3, tv[3] * coef);
    }
}

// ---------------------------------------------------------------------------
// Fused BatchNorm(eval) + LeakyReLU, in place
// ---------------------------------------------------------------------------
__global__ void gcn_bn_leaky(float* __restrict__ h, const float* __restrict__ g,
                             const float* __restrict__ be, const float* __restrict__ m,
                             const float* __restrict__ v, int n) {
    int gid  = blockIdx.x * blockDim.x + threadIdx.x;
    int node = gid >> 5;
    int c    = (gid & 31) << 2;
    if (node < n) {
        v4f gv = *(const v4f*)(g + c);
        v4f bv = *(const v4f*)(be + c);
        v4f mv = *(const v4f*)(m + c);
        v4f vv = *(const v4f*)(v + c);
        v4f y  = *(const v4f*)(h + (size_t)node * H_DIM + c);
#pragma unroll
        for (int k = 0; k < 4; ++k) {
            float sc = gv[k] * rsqrtf(vv[k] + BN_EPS);
            float z  = (y[k] - mv[k]) * sc + bv[k];
            y[k] = z > 0.0f ? z : LK_SLOPE * z;
        }
        *(v4f*)(h + (size_t)node * H_DIM + c) = y;
    }
}

// ---------------------------------------------------------------------------
// global_mean_pool
// ---------------------------------------------------------------------------
__global__ void pool_zero(float* __restrict__ sums, float* __restrict__ cnt, int G) {
    int i = blockIdx.x * blockDim.x + threadIdx.x;
    if (i < G * H_DIM) sums[i] = 0.0f;
    if (i < G) cnt[i] = 0.0f;
}

__global__ void pool_accum(const float* __restrict__ h, const int* __restrict__ batch,
                           float* __restrict__ sums, float* __restrict__ cnt, int n) {
    int gid  = blockIdx.x * blockDim.x + threadIdx.x;
    int node = gid >> 5;
    int c    = (gid & 31) << 2;
    if (node < n) {
        int g = batch[node];
        v4f hv = *(const v4f*)(h + (size_t)node * H_DIM + c);
        float* sp = sums + (size_t)g * H_DIM + c;
        atomicAdd(sp + 0, hv[0]);
        atomicAdd(sp + 1, hv[1]);
        atomicAdd(sp + 2, hv[2]);
        atomicAdd(sp + 3, hv[3]);
        if ((gid & 31) == 0) atomicAdd(&cnt[g], 1.0f);
    }
}

__global__ void pool_div(const float* __restrict__ sums, const float* __restrict__ cnt,
                         float* __restrict__ out, int G) {
    int i = blockIdx.x * blockDim.x + threadIdx.x;
    if (i < G * H_DIM) out[i] = sums[i] / fmaxf(cnt[i >> 7], 1.0f);
}

// ---------------------------------------------------------------------------
extern "C" void kernel_launch(void* const* d_in, const int* in_sizes, int n_in,
                              void* d_out, int out_size, void* d_ws, size_t ws_size,
                              hipStream_t stream) {
    const float* x   = (const float*)d_in[0];
    const float* ew  = (const float*)d_in[1];
    const float* W0  = (const float*)d_in[2];
    const float* b0  = (const float*)d_in[3];
    const float* W1  = (const float*)d_in[4];
    const float* b1  = (const float*)d_in[5];
    const float* W2  = (const float*)d_in[6];
    const float* b2  = (const float*)d_in[7];
    const float* g0  = (const float*)d_in[8];
    const float* be0 = (const float*)d_in[9];
    const float* m0  = (const float*)d_in[10];
    const float* v0  = (const float*)d_in[11];
    const float* g1  = (const float*)d_in[12];
    const float* be1 = (const float*)d_in[13];
    const float* m1  = (const float*)d_in[14];
    const float* v1  = (const float*)d_in[15];
    const int*   ei    = (const int*)d_in[16];
    const int*   batch = (const int*)d_in[17];

    const int N = in_sizes[0] / H_DIM;
    const int E = in_sizes[1];
    const int G = out_size / H_DIM;

    // workspace carve-up
    char* ws = (char*)d_ws;
    size_t off = 0;
    auto carve = [&](size_t bytes) { void* p = ws + off; off += (bytes + 255) & ~(size_t)255; return p; };
    float*     t    = (float*)carve((size_t)N * H_DIM * 4);
    float*     hA   = (float*)carve((size_t)N * H_DIM * 4);
    float*     hB   = (float*)carve((size_t)N * H_DIM * 4);
    float*     dinv = (float*)carve((size_t)N * 4);
    _Float16*  W0h  = (_Float16*)carve((size_t)H_DIM * H_DIM * 2);
    _Float16*  W1h  = (_Float16*)carve((size_t)H_DIM * H_DIM * 2);
    _Float16*  W2h  = (_Float16*)carve((size_t)H_DIM * H_DIM * 2);
    float*     sums = (float*)carve((size_t)G * H_DIM * 4);
    float*     cnt  = (float*)carve((size_t)G * 4);
    (void)ws_size;

    const int TPB = 256;
    // 1) degree^{-1/2}
    gcn_deg_init <<<(N + TPB - 1) / TPB, TPB, 0, stream>>>(dinv, N);
    gcn_deg_accum<<<(E + TPB - 1) / TPB, TPB, 0, stream>>>(ei, ew, dinv, E);
    gcn_deg_rsqrt<<<(N + TPB - 1) / TPB, TPB, 0, stream>>>(dinv, N);

    // 2) weight conversion (f32 [in][out] -> f16 [out][in])
    gcn_wcvt<<<(H_DIM * H_DIM + TPB - 1) / TPB, TPB, 0, stream>>>(W0, W0h);
    gcn_wcvt<<<(H_DIM * H_DIM + TPB - 1) / TPB, TPB, 0, stream>>>(W1, W1h);
    gcn_wcvt<<<(H_DIM * H_DIM + TPB - 1) / TPB, TPB, 0, stream>>>(W2, W2h);

    const int gemmBlocks = (N + 63) / 64;
    const int nodeVec    = (N * 32 + TPB - 1) / TPB;   // 32 threads/node
    const int edgeVec    = (E * 32 + TPB - 1) / TPB;   // 32 threads/edge

    auto layer = [&](const float* act, const _Float16* Wh, const float* bias, float* outb,
                     const float* g, const float* be, const float* m, const float* v) {
        gcn_gemm_wmma    <<<gemmBlocks, 128, 0, stream>>>(act, Wh, t, N);
        gcn_selfloop_bias<<<nodeVec, TPB, 0, stream>>>(t, dinv, bias, outb, N);
        gcn_scatter      <<<edgeVec, TPB, 0, stream>>>(ei, ew, dinv, t, outb, E);
        if (g) gcn_bn_leaky<<<nodeVec, TPB, 0, stream>>>(outb, g, be, m, v, N);
    };

    layer(x,  W0h, b0, hA, g0, be0, m0, v0);
    layer(hA, W1h, b1, hB, g1, be1, m1, v1);
    layer(hB, W2h, b2, hA, nullptr, nullptr, nullptr, nullptr);

    // 4) global mean pool
    pool_zero <<<(G * H_DIM + TPB - 1) / TPB, TPB, 0, stream>>>(sums, cnt, G);
    pool_accum<<<nodeVec, TPB, 0, stream>>>(hA, batch, sums, cnt, N);
    pool_div  <<<(G * H_DIM + TPB - 1) / TPB, TPB, 0, stream>>>(sums, cnt, (float*)d_out, G);
}
